// LIFConv2dLayer_45526653337818
// MI455X (gfx1250) — matrix operand order repaired and checked by reading, not compiled, not run on hardware
//
#include <hip/hip_runtime.h>

typedef __attribute__((ext_vector_type(16))) _Float16 v16h;
typedef __attribute__((ext_vector_type(8)))  _Float16 v8h;
typedef __attribute__((ext_vector_type(8)))  float    v8f;

namespace {
constexpr float kAlpha     = 0.90483741803596f;  // exp(-1e-3/0.01)
constexpr float kBeta      = 0.81873075307798f;  // exp(-1e-3/0.005)
constexpr float kGamma     = 0.67032004603564f;  // exp(-1e-3/0.0025)
constexpr float kTauMemEff = 10.508331944f;      // 1/(1-alpha)
constexpr float kTauSynEff = 5.5166559f;         // 1/(1-beta)
constexpr float kThr       = 1.0f;

constexpr int kT = 20, kB = 32, kC = 64, kH = 32, kW = 32, kOC = 128;
constexpr int kPadW = 34;
constexpr int kPadImg = kPadW * kPadW * kC;                 // 73984 f16 / image
constexpr size_t kSStride = (size_t)kB * kOC * kH * kW;     // 4194304 elems / t

constexpr int kPs = 72;                                     // LDS pixel stride (f16)

// workspace byte offsets (hipMalloc base is 256B aligned; all offsets 16B aligned)
constexpr size_t oWq = 0;                                   // f16 [9][128][64] = 147456 B
constexpr size_t oBq = 147456;                              // f32 [128]       = 512 B
constexpr size_t oP  = 147968;                              // f32 [B][C][H][W] = 8 MiB
constexpr size_t oQ  = oP + 8388608;
constexpr size_t oR  = oQ + 8388608;                        // f32 [B][OC][H][W] = 16 MiB
constexpr size_t oPn = oR + 16777216;                       // f16 [B][34][34][64]
}

__device__ __forceinline__ float quant8(float v) {
  float wc = fminf(fmaxf(v, -1.0f + 1.0f / 128.0f), 1.0f - 1.0f / 128.0f);
  return rintf(wc * 128.0f) * (1.0f / 128.0f);  // RNE matches jnp.round
}

// ---- weight/bias quantization, wq laid out [tap p][oc][c] for contiguous A rows ----
__global__ __launch_bounds__(256) void snn_quant_kernel(
    const float* __restrict__ w, const float* __restrict__ bias,
    _Float16* __restrict__ wq, float* __restrict__ bq) {
  int idx = blockIdx.x * 256 + threadIdx.x;
  if (idx < 9 * kOC * kC) {
    int p  = idx / (kOC * kC);
    int r  = idx - p * (kOC * kC);
    int oc = r / kC;
    int c  = r - oc * kC;
    int ky = p / 3, kx = p - 3 * ky;
    float v = w[(((size_t)oc * kC + c) * 3 + ky) * 3 + kx];  // [OC][C][3][3]
    wq[idx] = (_Float16)quant8(v);
  }
  if (idx < kOC) bq[idx] = quant8(bias[idx]);
}

// ---- P/Q LIF update; writes Pn as f16 into zero-padded NHWC buffer ----
__global__ __launch_bounds__(256) void snn_lif_state_kernel(
    const float* __restrict__ xt, float* __restrict__ P, float* __restrict__ Q,
    _Float16* __restrict__ pnpad, int first) {
  int idx = blockIdx.x * 256 + threadIdx.x;   // flat over [B][34][34][64]
  int c = idx & 63;
  int r = idx >> 6;
  int px = r % kPadW; r /= kPadW;
  int py = r % kPadW;
  int b  = r / kPadW;
  if (b >= kB) return;
  if (px == 0 || px == kPadW - 1 || py == 0 || py == kPadW - 1) {
    pnpad[idx] = (_Float16)0.0f;
    return;
  }
  int y = py - 1, x = px - 1;
  size_t s = (((size_t)b * kC + c) * kH + y) * kW + x;   // NCHW state index
  float pn, qn;
  if (first) {
    pn = 0.0f;
    qn = kTauSynEff * xt[s];
  } else {
    float po = P[s], qo = Q[s];
    pn = kAlpha * po + kTauMemEff * qo;   // uses previous-step P,Q
    qn = kBeta  * qo + kTauSynEff * xt[s];
  }
  P[s] = pn;
  Q[s] = qn;
  pnpad[idx] = (_Float16)pn;
}

// ---- implicit-GEMM 3x3 conv via WMMA f16 + fused R update / threshold ----
// grid: 512 blocks = 32 images x 16 row-pairs; 256 threads = 8 waves.
// Wave w computes OC rows [16w,16w+16) x 64 pixels (four 16x16 f32 accums).
// The 4-row x 34-col x 64-ch input patch is staged in LDS ONCE via gfx1250
// async global->LDS copies (no VGPR round-trip, ASYNCcnt-tracked); all 18
// K-chunks (9 taps x 2 channel halves) read shifted windows of it.
__global__ __launch_bounds__(256) void snn_conv_lif_wmma_kernel(
    const _Float16* __restrict__ pnpad, const _Float16* __restrict__ wq,
    const float* __restrict__ bq, float* __restrict__ R,
    const float* __restrict__ sprev, float* __restrict__ sout, int first) {
  __shared__ _Float16 patch[136 * kPs];  // [4 rows][34 px][64 ch], px stride 72

  const int tid   = threadIdx.x;
  const int lane  = tid & 31;
  const int wave  = tid >> 5;
  const int b     = blockIdx.x >> 4;
  const int ytile = blockIdx.x & 15;

  const int col = lane & 15;   // N column / A row within 16
  const int hi  = lane >> 4;   // 0: K-low half lanes, 1: K-high half lanes

  const _Float16* pimg = pnpad + (size_t)b * kPadImg;

  // ---- async patch staging: 136 pixels x 8 chunks of 8 f16 = 1088 b128 copies.
  // generic LDS pointer = {shared_aperture, lds_offset}; low 32 bits are the
  // DS-unit byte address, which is what the async instruction's VDST expects.
  for (int i = tid; i < 136 * 8; i += 256) {
    const int pix = i >> 3;              // 0..135 = row*34 + px
    const int co  = (i & 7) * 8;         // channel sub-offset
    const int row = pix / 34;            // 0..3 (padded rows 2*ytile + row)
    const int px  = pix - row * 34;      // 0..33
    const _Float16* src =
        pimg + ((size_t)(2 * ytile + row) * kPadW + px) * kC + co;
    unsigned ldsa = (unsigned)(size_t)&patch[pix * kPs + co];
    asm volatile("global_load_async_to_lds_b128 %0, %1, off"
                 :
                 : "v"(ldsa), "v"(src)
                 : "memory");
  }
  asm volatile("s_wait_asynccnt 0x0" ::: "memory");
  __syncthreads();

  v8f acc0 = {}, acc1 = {}, acc2 = {}, acc3 = {};

  for (int p = 0; p < 9; ++p) {
    const int ky = p / 3;
    const int kx = p - 3 * ky;
#pragma unroll
    for (int ch = 0; ch < 2; ++ch) {     // channel halves of 32 (K chunks)
      // A fragment (16x32): lane holds row M=col; elems[0..15]=K base+0..15,
      // base = hi*16  (contiguous K order, matching B)
      v16h afrag;
      {
        const _Float16* wb =
            wq + (((size_t)p * kOC) + 16 * wave + col) * kC + ch * 32 + hi * 16;
        v8h* ah = (v8h*)&afrag;
        ah[0] = *(const v8h*)wb;
        ah[1] = *(const v8h*)(wb + 8);
      }
      // B fragments (32x16): lane = column `col`, K range = hi*16 .. +15.
      // Subtile s covers pixels 16s..16s+15: local row = s>>1, x = (s&1)*16+col.
      v16h bf0, bf1, bf2, bf3;
      {
        const int cbase = ch * 32 + hi * 16;
        const _Float16* bp;
        v8h* bh;
        bp = &patch[((0 + ky) * 34 + (0  + col + kx)) * kPs + cbase];
        bh = (v8h*)&bf0; bh[0] = *(const v8h*)bp; bh[1] = *(const v8h*)(bp + 8);
        bp = &patch[((0 + ky) * 34 + (16 + col + kx)) * kPs + cbase];
        bh = (v8h*)&bf1; bh[0] = *(const v8h*)bp; bh[1] = *(const v8h*)(bp + 8);
        bp = &patch[((1 + ky) * 34 + (0  + col + kx)) * kPs + cbase];
        bh = (v8h*)&bf2; bh[0] = *(const v8h*)bp; bh[1] = *(const v8h*)(bp + 8);
        bp = &patch[((1 + ky) * 34 + (16 + col + kx)) * kPs + cbase];
        bh = (v8h*)&bf3; bh[0] = *(const v8h*)bp; bh[1] = *(const v8h*)(bp + 8);
      }
      acc0 = __builtin_amdgcn_wmma_f32_16x16x32_f16(false, afrag, false, bf0, (short)0, acc0, false, false);
      acc1 = __builtin_amdgcn_wmma_f32_16x16x32_f16(false, afrag, false, bf1, (short)0, acc1, false, false);
      acc2 = __builtin_amdgcn_wmma_f32_16x16x32_f16(false, afrag, false, bf2, (short)0, acc2, false, false);
      acc3 = __builtin_amdgcn_wmma_f32_16x16x32_f16(false, afrag, false, bf3, (short)0, acc3, false, false);
    }
  }

  // epilogue: U = conv + bias + Rn ; Rn = gamma*R - thr*S_prev ; S = (U > thr)
  const int y0 = 2 * ytile;
  float* outb      = sout  + (size_t)b * kOC * (kH * kW);
  const float* spb = sprev + (size_t)b * kOC * (kH * kW);
  float* rb        = R     + (size_t)b * kOC * (kH * kW);

  auto epilog = [&](int s, v8f a) {
    const int pix = 16 * s + col;
    const int y = y0 + (pix >> 5);
    const int x = pix & 31;
#pragma unroll
    for (int r = 0; r < 8; ++r) {
      // C/D layout: VGPR r holds M=r (lanes 0-15) and M=r+8 (lanes 16-31)
      const int oc = 16 * wave + r + 8 * hi;
      const size_t idx = (size_t)oc * (kH * kW) + y * kW + x;
      float rn = first ? 0.0f : (kGamma * rb[idx] - kThr * spb[idx]);
      float u  = a[r] + bq[oc] + rn;
      rb[idx]   = rn;
      outb[idx] = (u > kThr) ? 1.0f : 0.0f;
    }
  };
  epilog(0, acc0);
  epilog(1, acc1);
  epilog(2, acc2);
  epilog(3, acc3);
}

// ---- maxpool(2x2) on the fly + local readout: ro[b,o] = sum_f pooled*w_ro ----
__global__ __launch_bounds__(256) void snn_readout_kernel(
    const float* __restrict__ S, const float* __restrict__ wro,
    float* __restrict__ ro) {
  const int b = blockIdx.x / 10;
  const int o = blockIdx.x - 10 * b;
  const float* Sb = S + (size_t)b * kOC * (kH * kW);
  const float* wr = wro + (size_t)o * (kOC * 16 * 16);
  float sum = 0.0f;
  for (int f = threadIdx.x; f < kOC * 16 * 16; f += 256) {
    const int pw = f & 15;
    const int ph = (f >> 4) & 15;
    const int c  = f >> 8;
    const float* sp = Sb + (size_t)c * (kH * kW) + (2 * ph) * kW + 2 * pw;
    float m = fmaxf(fmaxf(sp[0], sp[1]), fmaxf(sp[kW], sp[kW + 1]));
    sum += m * wr[f];
  }
  __shared__ float red[256];
  red[threadIdx.x] = sum;
  __syncthreads();
  for (int off = 128; off > 0; off >>= 1) {
    if (threadIdx.x < off) red[threadIdx.x] += red[threadIdx.x + off];
    __syncthreads();
  }
  if (threadIdx.x == 0) ro[b * 10 + o] = red[0];
}

extern "C" void kernel_launch(void* const* d_in, const int* in_sizes, int n_in,
                              void* d_out, int out_size, void* d_ws, size_t ws_size,
                              hipStream_t stream) {
  (void)in_sizes; (void)n_in; (void)out_size; (void)ws_size;
  const float* x    = (const float*)d_in[0];   // [T,B,C,H,W]
  const float* w    = (const float*)d_in[1];   // [OC,C,3,3]
  const float* bias = (const float*)d_in[2];   // [OC]
  const float* w_ro = (const float*)d_in[3];   // [10, OC*16*16]
  float* out = (float*)d_out;                  // S_seq flat, then ro_seq flat
  char* ws = (char*)d_ws;

  _Float16* wq    = (_Float16*)(ws + oWq);
  float*    bq    = (float*)(ws + oBq);
  float*    P     = (float*)(ws + oP);
  float*    Q     = (float*)(ws + oQ);
  float*    R     = (float*)(ws + oR);
  _Float16* pnpad = (_Float16*)(ws + oPn);

  snn_quant_kernel<<<(9 * kOC * kC + 255) / 256, 256, 0, stream>>>(w, bias, wq, bq);

  float* ro_base = out + (size_t)kT * kSStride;
  const int lifBlocks = (kB * kPadImg) / 256;        // 9248, exact
  for (int t = 0; t < kT; ++t) {
    const float* xt = x + (size_t)t * kB * kC * kH * kW;
    snn_lif_state_kernel<<<lifBlocks, 256, 0, stream>>>(xt, P, Q, pnpad, t == 0 ? 1 : 0);

    float* sout = out + (size_t)t * kSStride;
    const float* sprev = (t == 0) ? sout : (sout - kSStride);  // unread when first=1
    snn_conv_lif_wmma_kernel<<<kB * 16, 256, 0, stream>>>(pnpad, wq, bq, R, sprev, sout,
                                                          t == 0 ? 1 : 0);

    snn_readout_kernel<<<kB * 10, 256, 0, stream>>>(sout, w_ro, ro_base + (size_t)t * kB * 10);
  }
}
